// CFConv_9715216023986
// MI455X (gfx1250) — compile-verified, matrix operand order checked by reading, not run on hardware
//
#include <hip/hip_runtime.h>

typedef __attribute__((ext_vector_type(16))) _Float16 v16h;
typedef __attribute__((ext_vector_type(8)))  _Float16 v8h;
typedef __attribute__((ext_vector_type(8)))  float    v8f;
typedef __attribute__((ext_vector_type(4)))  unsigned int v4u;
typedef __attribute__((ext_vector_type(8)))  int      v8i;
typedef __attribute__((ext_vector_type(4)))  int      v4i;

constexpr int FD  = 128;   // feature dim F
constexpr int GD  = 25;    // gaussian dim G
constexpr int GP  = 32;    // G padded to WMMA K
constexpr int NN  = 64;    // neighbors
constexpr int NAT = 512;   // atoms per batch
constexpr int BNA = 16 * 512;

// shifted softplus: ln(1+e^x) - ln2, base-2 domain on raw HW trans ops.
__device__ __forceinline__ float sspf(float x) {
  float t = __builtin_amdgcn_exp2f(-1.4426950408889634f * fabsf(x));
  float l = __builtin_amdgcn_logf(1.0f + t);
  return fmaf(0.6931471805599453f, l - 1.0f, fmaxf(x, 0.0f));
}

// TDM: DMA a flat run of f16 from global to LDS (ISA 8.3/8.4 D# layout).
// 1-D tile: tensor_dim0 = tile_dim0 = nhalfs, data_size = 2B, type = 2.
__device__ __forceinline__ void tdm_load_lds(void* lds_dst, const void* gsrc,
                                             int nhalfs) {
  uint64_t ga = (uint64_t)(uintptr_t)gsrc;
  v4u g0;
  g0[0] = 1u;                                         // count=1 (valid D#)
  g0[1] = (unsigned)(uintptr_t)lds_dst;               // lds_addr (bytes)
  g0[2] = (unsigned)ga;                               // global_addr[31:0]
  g0[3] = ((unsigned)(ga >> 32) & 0x01FFFFFFu) | (2u << 30); // [56:32]+type=2
  v8i g1 = (v8i)0;
  g1[0] = 1 << 16;                                    // data_size: 1 => 2B
  g1[1] = (nhalfs & 0xFFFF) << 16;                    // tensor_dim0[15:0]
  g1[2] = (nhalfs >> 16) & 0xFFFF;                    // tensor_dim0[31:16]
  g1[3] = (nhalfs & 0xFFFF) << 16;                    // tile_dim0
  v4i gz4 = (v4i)0;
  v8i gz8 = (v8i)0;
  __builtin_amdgcn_tensor_load_to_lds(g0, g1, gz4, gz4, gz8, 0);
}

// XOR-swizzled LDS index for f16 tiles with row stride S halfs (S/8 chunks
// of 16B per row): 16 consecutive rows hit 16 distinct 4-bank groups.
template <int S>
__device__ __forceinline__ int swz(int row, int k) {
  constexpr int NCH = S / 8;
  return row * S + ((((k >> 3) ^ row) & (NCH - 1)) << 3) + (k & 7);
}

// 16x32 f16 WMMA A-fragment (or B-fragment of transposed tile) from swizzled
// LDS. ISA 7.12.2: lanes 0-15 hold row=lane, K {k0..k0+7, k0+16..k0+23};
// lanes 16-31 K shifted by +8.
template <int S>
__device__ __forceinline__ v16h ld_frag(const _Float16* base, int row, int k0) {
  const int lane = threadIdx.x & 31;
  const int k = k0 + ((lane >> 4) << 3);
  v8h lo = *(const v8h*)(base + swz<S>(row, k));
  v8h hi = *(const v8h*)(base + swz<S>(row, k + 16));
  v16h r;
#pragma unroll
  for (int i = 0; i < 8; ++i) { r[i] = lo[i]; r[i + 8] = hi[i]; }
  return r;
}

// One-shot: bake f16, transposed ([n][k]), swizzled LDS images of the weight
// matrices into workspace. GEMM workgroups then TDM-copy them into LDS.
__global__ __launch_bounds__(256) void prep_weights(
    const float* __restrict__ Win2f, const float* __restrict__ Wf1,
    const float* __restrict__ Wf2,   const float* __restrict__ Wout,
    _Float16* __restrict__ win2f_img, _Float16* __restrict__ wf1_img,
    _Float16* __restrict__ wf2_img,   _Float16* __restrict__ wout_img) {
  int t = blockIdx.x * 256 + threadIdx.x;
  if (t < FD * FD) {
    int k = t >> 7, n = t & 127;          // source is [k][n] row-major
    int d = swz<FD>(n, k);
    win2f_img[d] = (_Float16)Win2f[t];
    wf2_img[d]   = (_Float16)Wf2[t];
    wout_img[d]  = (_Float16)Wout[t];
  }
  if (t < FD * GP) {
    int f = t >> 5, g = t & 31;
    wf1_img[swz<GP>(f, g)] = (_Float16)((g < GD) ? Wf1[g * FD + f] : 0.0f);
  }
}

// C = A[M,128] @ W[128,128] (+bias, +ssp if ACT). One block = 64 rows.
// Computed transposed (D' = W^T x A^T): wave owns a 16-wide f-strip, lanes own
// rows m; each lane's 8 outputs are contiguous f -> vector stores.
template <bool ACT>
__global__ __launch_bounds__(256) void gemm128(const float* __restrict__ A,
                                               const _Float16* __restrict__ Wimg,
                                               const float* __restrict__ bias,
                                               float* __restrict__ C) {
  __shared__ _Float16 sA[64 * FD];     // 16 KB, swizzled [m][k]
  __shared__ _Float16 sW[FD * FD];     // 32 KB, pre-swizzled image (TDM fill)
  const int tid  = threadIdx.x;
  const int row0 = blockIdx.x * 64;

  if (tid == 0) tdm_load_lds(sW, Wimg, FD * FD);      // async L2 -> LDS
  for (int i = tid; i < 64 * FD; i += 256) {
    int n = i >> 7, k = i & 127;
    sA[swz<FD>(n, k)] = (_Float16)A[(size_t)row0 * FD + i];
  }
  __builtin_amdgcn_s_wait_tensorcnt(0);
  __syncthreads();

  const int lane = tid & 31, wave = tid >> 5;
  const int fbase = wave * 16;                       // f-strip of this wave
  const int fb    = fbase + ((lane >> 4) << 3);      // lane's 8-f group

  float bv[8];
  if constexpr (ACT) {
#pragma unroll
    for (int r = 0; r < 8; ++r) bv[r] = bias[fb + r];
  }

#pragma unroll
  for (int mt = 0; mt < 4; ++mt) {
    v8f acc = {};
#pragma unroll
    for (int k0 = 0; k0 < FD; k0 += 32) {
      v16h aW = ld_frag<FD>(sW, fbase + (lane & 15), k0);
      v16h bA = ld_frag<FD>(sA, mt * 16 + (lane & 15), k0);
      acc = __builtin_amdgcn_wmma_f32_16x16x32_f16(
          false, aW, false, bA, (short)0, acc, false, false);
    }
    const int m = mt * 16 + (lane & 15);
    v8f vr;
#pragma unroll
    for (int r = 0; r < 8; ++r) {
      float v = acc[r];
      if constexpr (ACT) v = sspf(v + bv[r]);
      vr[r] = v;
    }
    *(v8f*)(C + (size_t)(row0 + m) * FD + fb) = vr;  // 2x global_store_b128
  }
}

// Fused filter-network + gather + masked aggregation. One block per atom.
__global__ __launch_bounds__(256) void cfconv_core(
    const float* __restrict__ fij,        // [B*Na, NN, G]
    const int*   __restrict__ nbh,        // [B*Na, NN]
    const float* __restrict__ mask,       // [B*Na, NN]
    const float* __restrict__ y,          // [B, Na, F] (L2-resident)
    const _Float16* __restrict__ wf1_img, // pre-swizzled [FD*GP]
    const float* __restrict__ bf1,        // [F]
    const _Float16* __restrict__ wf2_img, // pre-swizzled [FD*FD]
    const float* __restrict__ bf2,        // [F]
    float* __restrict__ agg)              // [B*Na, F]
{
  __shared__ _Float16 sA1[NN * GP];     // 4 KB   f_ij tile, swizzled [m][g]
  __shared__ _Float16 sW1[FD * GP];     // 8 KB   image (TDM fill)  [f][g]
  __shared__ _Float16 sT1[NN * FD];     // 16 KB  ssp(GEMM1) f16, swizzled [m][f]
  __shared__ _Float16 sW2[FD * FD];     // 32 KB  image (TDM fill)  [f][k]
  __shared__ int   sN[NN];
  __shared__ float sM[NN];
  __shared__ float sAgg[FD];

  const int tid  = threadIdx.x;
  const int atom = blockIdx.x;
  const int b    = atom >> 9;                       // Na = 512
  const float* yb = y + (size_t)b * NAT * FD;
  const float* fa = fij + (size_t)atom * NN * GD;

  if (tid == 0) {
    tdm_load_lds(sW1, wf1_img, FD * GP);            // async L2 -> LDS
    tdm_load_lds(sW2, wf2_img, FD * FD);
  }
  for (int i = tid; i < NN * GP; i += 256) {
    int n = i >> 5, g = i & 31;
    sA1[swz<GP>(n, g)] = (_Float16)((g < GD) ? fa[n * GD + g] : 0.0f);
  }
  if (tid < NN) {
    sN[tid] = nbh[(size_t)atom * NN + tid];
    sM[tid] = mask[(size_t)atom * NN + tid];
  }
  if (tid < FD) sAgg[tid] = 0.0f;
  __builtin_amdgcn_s_wait_tensorcnt(0);
  __syncthreads();

  const int lane = tid & 31, wave = tid >> 5;

  // GEMM1 transposed: T1^T = W1^T @ A1^T. Wave owns f-strip [wave*16, +16);
  // lane's 8 outputs are contiguous f of one row m -> one b128 store.
  {
    const int fbase = wave * 16;
    const int fb    = fbase + ((lane >> 4) << 3);
    float bv[8];
#pragma unroll
    for (int r = 0; r < 8; ++r) bv[r] = bf1[fb + r];
    v16h aW = ld_frag<GP>(sW1, fbase + (lane & 15), 0);
#pragma unroll
    for (int mt = 0; mt < 4; ++mt) {
      v16h bA = ld_frag<GP>(sA1, mt * 16 + (lane & 15), 0);
      v8f acc = {};
      acc = __builtin_amdgcn_wmma_f32_16x16x32_f16(
          false, aW, false, bA, (short)0, acc, false, false);
      const int m = mt * 16 + (lane & 15);
      v8h pk;
#pragma unroll
      for (int r = 0; r < 8; ++r) pk[r] = (_Float16)sspf(acc[r] + bv[r]);
      *(v8h*)(sT1 + swz<FD>(m, fb)) = pk;           // one ds_store_b128
    }
  }
  __syncthreads();

  // GEMM2: [64,128] @ [128,128] with m on VGPR axis (wanted by the reduce).
  const int m16 = (wave & 3) * 16;
  const int h   = (wave >> 2) * 64;
  v8f acc2[4] = {};
#pragma unroll
  for (int k0 = 0; k0 < FD; k0 += 32) {
    v16h a = ld_frag<FD>(sT1, m16 + (lane & 15), k0);
#pragma unroll
    for (int nt = 0; nt < 4; ++nt) {
      v16h bb = ld_frag<FD>(sW2, h + nt * 16 + (lane & 15), k0);
      acc2[nt] = __builtin_amdgcn_wmma_f32_16x16x32_f16(
          false, a, false, bb, (short)0, acc2[nt], false, false);
    }
  }

  // Epilogue: (W + b_f2) * y[nbh[m]] * mask[m], reduce over neighbors.
#pragma unroll
  for (int nt = 0; nt < 4; ++nt) {
    int f = h + nt * 16 + (lane & 15);
    float bv = bf2[f];
    float part = 0.0f;
#pragma unroll
    for (int r = 0; r < 8; ++r) {
      int m = m16 + r + ((lane >> 4) << 3);
      float g = yb[(size_t)sN[m] * FD + f];   // coalesced 16-lane rows
      part += (acc2[nt][r] + bv) * g * sM[m];
    }
    atomicAdd(&sAgg[f], part);                 // ds_add_f32
  }
  __syncthreads();
  if (tid < FD) agg[(size_t)atom * FD + tid] = sAgg[tid];
}

extern "C" void kernel_launch(void* const* d_in, const int* in_sizes, int n_in,
                              void* d_out, int out_size, void* d_ws, size_t ws_size,
                              hipStream_t stream) {
  const float* x     = (const float*)d_in[0];
  // d_in[1] = r_ij (unused by the reference math)
  const float* fij   = (const float*)d_in[2];
  const int*   nbh   = (const int*)d_in[3];
  const float* mask  = (const float*)d_in[4];
  const float* Win2f = (const float*)d_in[5];
  const float* Wf1   = (const float*)d_in[6];
  const float* bf1   = (const float*)d_in[7];
  const float* Wf2   = (const float*)d_in[8];
  const float* bf2   = (const float*)d_in[9];
  const float* Wout  = (const float*)d_in[10];
  const float* bout  = (const float*)d_in[11];
  float* out = (float*)d_out;

  float*    yws   = (float*)d_ws;                    // [8192,128] f32, 4 MB
  float*    aggw  = yws + (size_t)BNA * FD;          // [8192,128] f32, 4 MB
  _Float16* wimg  = (_Float16*)(aggw + (size_t)BNA * FD);
  _Float16* win2f_img = wimg;                        // 16384 halfs
  _Float16* wf1_img   = wimg + FD * FD;              //  4096 halfs
  _Float16* wf2_img   = wf1_img + FD * GP;           // 16384 halfs
  _Float16* wout_img  = wf2_img + FD * FD;           // 16384 halfs

  prep_weights<<<64, 256, 0, stream>>>(Win2f, Wf1, Wf2, Wout,
                                       win2f_img, wf1_img, wf2_img, wout_img);
  gemm128<false><<<BNA / 64, 256, 0, stream>>>(x, win2f_img, nullptr, yws);
  cfconv_core<<<BNA, 256, 0, stream>>>(fij, nbh, mask, yws,
                                       wf1_img, bf1, wf2_img, bf2, aggw);
  gemm128<true><<<BNA / 64, 256, 0, stream>>>(aggw, wout_img, bout, out);
}
